// Despawn2D_18485539242199
// MI455X (gfx1250) — matrix operand order repaired and checked by reading, not compiled, not run on hardware
//
#include <hip/hip_runtime.h>
#include <stdint.h>

// ---------------------------------------------------------------------------
// DeSpaWN fused forward for MI455X (gfx1250, wave32).
//   - one workgroup per row; row + all coefficients LDS-resident (96 KB/WG)
//   - global -> LDS row load via CDNA5 async-to-LDS DMA path
//   - 8-level DB2 analysis as a v_wmma_f32_16x16x32_bf16 filterbank
//     (A = 16 position-tiles x 32-wide even-aligned window -> ds_load_b64
//      pairs; B = banded filter matrix, cols 0-7 = h, cols 8-15 = g)
//   - full tiles take a branchless store path (no per-lane EXEC juggling)
//   - synthesis cascade in fp32 VALU while coeffs stream back to HBM
//     via global_store_async_from_lds_b128 (overlap compute with stores)
// ---------------------------------------------------------------------------

typedef __attribute__((ext_vector_type(16))) __bf16 v16bf;
typedef __attribute__((ext_vector_type(8)))  float  v8f;

#define LEVELS   8
#define ROWLEN   8192
#define THREADS  256
#define NWAVES   (THREADS / 32)

__device__ __forceinline__ unsigned lds_off(const void* p) {
  // flat LDS address: low 32 bits are the wave-relative LDS byte offset
  return (unsigned)(uintptr_t)p;
}

__device__ __forceinline__ void async_load_b128(const void* gaddr, void* lds) {
  asm volatile("global_load_async_to_lds_b128 %0, %1, off"
               :: "v"(lds_off(lds)), "v"((unsigned long long)(uintptr_t)gaddr)
               : "memory");
}

__device__ __forceinline__ void async_store_b128(void* gaddr, const void* lds) {
  asm volatile("global_store_async_from_lds_b128 %0, %1, off"
               :: "v"((unsigned long long)(uintptr_t)gaddr), "v"(lds_off(lds))
               : "memory");
}

__device__ __forceinline__ void wait_async0() {
#if __has_builtin(__builtin_amdgcn_s_wait_asynccnt)
  __builtin_amdgcn_s_wait_asynccnt(0);
#else
  asm volatile("s_wait_asynccnt 0" ::: "memory");
#endif
}

__global__ __launch_bounds__(THREADS)
void despawn_fused(const float* __restrict__ x,
                   const float* __restrict__ scaling,
                   float* __restrict__ out_rec,
                   float* __restrict__ out_coef)
{
  extern __shared__ float smem[];
  float* sh_x = smem;                     // 8192: input row, later final rec
  float* sh_c = smem + ROWLEN;            // 8192: coeffs laid out as output
  float* sh_a = smem + 2 * ROWLEN;        // 4096: ping
  float* sh_b = smem + 2 * ROWLEN + 4096; // 4096: pong

  const int row  = blockIdx.x;
  const int tid  = threadIdx.x;
  const int lane = tid & 31;
  const int wave = tid >> 5;
  const int nCol = lane & 15;   // B/D column (N index)
  const int hsel = lane >> 4;   // lane-half select per WMMA layouts

  // ---------------- stage 1: async DMA row into LDS ----------------
  const float* gx = x + (size_t)row * ROWLEN;
  for (int u = tid; u < ROWLEN / 4; u += THREADS)
    async_load_b128(gx + 4 * u, sh_x + 4 * u);
  wait_async0();
  __syncthreads();

  // ---------------- stage 2: analysis cascade (WMMA filterbank) ----
  // approx[p] = sum_t h[t] * a[(2p - t) mod N]
  // detail[p] = sum_t g[t] * a[(2p - t) mod N],  g = [h3,-h2,h1,-h0]
  int N = ROWLEN;
  const float* src = sh_x;
  for (int lev = 0; lev < LEVELS; ++lev) {
    float h[4], g[4];
#pragma unroll
    for (int k = 0; k < 4; ++k) h[k] = scaling[4 * lev + k];
    g[0] = h[3]; g[1] = -h[2]; g[2] = h[1]; g[3] = -h[0];

    const int half = N >> 1;
    float* dst_a = (lev == LEVELS - 1) ? (sh_c + ROWLEN - 32)
                                       : ((lev & 1) ? sh_b : sh_a);
    float* dst_d = sh_c + (ROWLEN - (ROWLEN >> lev));
    // branchless per-lane destination: cols 0-7 -> approx, 8-15 -> detail
    float* base = (nCol < 8) ? dst_a : dst_d;

    // B (32x16 bf16): banded filter matrix, constant per level.
    // window starts at w = 2*p_m - 4 (even), so B[K][n] = f[2*(n&7)+4-K].
    // 16-bit B layout: VGPR0-3: K=(0..7)+8*hsel, VGPR4-7: K=(16..23)+8*hsel
    v16bf bm;
#pragma unroll
    for (int v = 0; v < 8; ++v) {
#pragma unroll
      for (int e = 0; e < 2; ++e) {
        const int K = ((v < 4) ? 2 * v : 8 + 2 * v) + e + 8 * hsel;
        const int t = 2 * (nCol & 7) + 4 - K;
        float coef = 0.f;
        if (t >= 0 && t <= 3) coef = (nCol < 8) ? h[t] : g[t];
        bm[2 * v + e] = (__bf16)coef;
      }
    }

    const float2* src2 = (const float2*)src;

    // Each WMMA tile covers 128 output positions (16 M-tiles x 8 positions),
    // producing both approx and detail. Tiles split across the 8 waves.
    const int ntiles = (half + 127) >> 7;
    for (int tIdx = wave; tIdx < ntiles; tIdx += NWAVES) {
      const int pbase = tIdx << 7;
      // even window start for this lane's M-tile (circular)
      const int w = 2 * (pbase + 8 * (lane & 15)) - 4;
      // 16-bit A layout 16x32: lane M=l&15; VGPR0-3: K=(0..7)+8*hsel,
      // VGPR4-7: K=(16..23)+8*hsel.  K-pairs are even-aligned -> b64 loads.
      v16bf am;
#pragma unroll
      for (int v = 0; v < 8; ++v) {
        const int K0 = ((v < 4) ? 2 * v : 8 + 2 * v) + 8 * hsel;
        const int e2 = ((w + K0 + N) & (N - 1)) >> 1;   // even -> pair index
        const float2 pr = src2[e2];
        am[2 * v]     = (__bf16)pr.x;
        am[2 * v + 1] = (__bf16)pr.y;
      }
      v8f acc = {};
      acc = __builtin_amdgcn_wmma_f32_16x16x32_bf16(
          false, am, false, bm, (short)0, acc, false, false);
      // D layout: VGPR r -> M = r + 8*hsel, N = nCol
      if (pbase + 128 <= half) {          // full tile: unconditional stores
#pragma unroll
        for (int r = 0; r < 8; ++r)
          base[pbase + 8 * (r + 8 * hsel) + (nCol & 7)] = acc[r];
      } else {                            // tail tile (levels 6/7 only)
#pragma unroll
        for (int r = 0; r < 8; ++r) {
          const int p = pbase + 8 * (r + 8 * hsel) + (nCol & 7);
          if (p < half) base[p] = acc[r];
        }
      }
    }
    __syncthreads();
    src = dst_a;
    N = half;
  }

  // ---------------- stage 3: stream coeffs to HBM (async, overlapped) ----
  float* gc = out_coef + (size_t)row * ROWLEN;
  for (int u = tid; u < ROWLEN / 4; u += THREADS)
    async_store_b128(gc + 4 * u, sh_c + 4 * u);

  // ---------------- stage 4: synthesis cascade (fp32 VALU) ----------------
  // rec_new[n] = sum_k g[k]*ud[(n+k) mod 2m] + h[k]*ua[(n+k) mod 2m]
  // with ud/ua the zero-insert upsamples of detail / previous rec.
  const float* rec = sh_c + ROWLEN - 32;  // final approx (len 32)
  for (int lev = LEVELS - 1; lev >= 0; --lev) {
    float h[4], g[4];
#pragma unroll
    for (int k = 0; k < 4; ++k) h[k] = scaling[4 * lev + k];
    g[0] = h[3]; g[1] = -h[2]; g[2] = h[1]; g[3] = -h[0];

    const int m = ROWLEN >> (lev + 1);          // detail length at this level
    const float* dc = sh_c + (ROWLEN - (ROWLEN >> lev));
    float* dst = (lev == 0) ? sh_x : ((lev & 1) ? sh_a : sh_b);

    for (int n = tid; n < 2 * m; n += THREADS) {
      float s = 0.f;
#pragma unroll
      for (int k = 0; k < 4; ++k) {
        const int j = (n + k) & (2 * m - 1);
        if (!(j & 1)) {                         // only even taps of upsampled
          const int jj = j >> 1;
          s = fmaf(g[k], dc[jj], s);
          s = fmaf(h[k], rec[jj], s);
        }
      }
      dst[n] = s;
    }
    __syncthreads();
    rec = dst;
  }

  // ---------------- stage 5: stream rec to HBM ----------------
  float* gr = out_rec + (size_t)row * ROWLEN;
  for (int u = tid; u < ROWLEN / 4; u += THREADS)
    async_store_b128(gr + 4 * u, sh_x + 4 * u);
  wait_async0();
}

extern "C" void kernel_launch(void* const* d_in, const int* in_sizes, int n_in,
                              void* d_out, int out_size, void* d_ws, size_t ws_size,
                              hipStream_t stream) {
  (void)n_in; (void)out_size; (void)d_ws; (void)ws_size;
  const float* x  = (const float*)d_in[0];
  const float* sc = (const float*)d_in[1];
  const int rows  = in_sizes[0] / ROWLEN;       // 2048
  float* rec  = (float*)d_out;                  // (rows, 8192)
  float* coef = (float*)d_out + (size_t)rows * ROWLEN;  // (rows, 8192)
  const size_t lds_bytes = (size_t)(2 * ROWLEN + 2 * 4096) * sizeof(float); // 96 KB
  despawn_fused<<<dim3(rows), dim3(THREADS), lds_bytes, stream>>>(x, sc, rec, coef);
}
